// QNetwork_89103391522965
// MI455X (gfx1250) — matrix operand order repaired and checked by reading, not compiled, not run on hardware
//
#include <hip/hip_runtime.h>
#include <hip/hip_bf16.h>

typedef __attribute__((ext_vector_type(16))) _Float16 v16h;
typedef __attribute__((ext_vector_type(8)))  _Float16 v8h;
typedef __attribute__((ext_vector_type(8)))  float    v8f;
typedef __attribute__((ext_vector_type(4)))  float    v4f;

// ---------------------------------------------------------------------------
// B=256 graphs, 64 nodes/graph, 1024 edges/graph, dims 128->256->128,
// FG=64, SO=128, NA=8, HA=256.  Segments are contiguous fixed-size blocks.
// ---------------------------------------------------------------------------

// ---------------------------------------------------------------------------
// Weight pre-swizzle: f32 [K,N] row-major  ->  f16 WMMA-B tile register order.
// B tile (K=32, N=16): lane l holds 16 halfs; n = nt*16 + (l&15),
// k = kt*32 + 8*(l>>4) + (j&7) + 16*(j>>3).  Linear: dst[((kt*NT + nt)*32 + lane)*16 + j]
// ---------------------------------------------------------------------------
__global__ __launch_bounds__(256) void k_convert(
    const float* __restrict__ We1, const float* __restrict__ Wn1,
    const float* __restrict__ We2, const float* __restrict__ Wn2,
    _Float16* __restrict__ We1h, _Float16* __restrict__ Wn1h,
    _Float16* __restrict__ We2h, _Float16* __restrict__ Wn2h) {
  int tid  = blockIdx.x * 256 + threadIdx.x;      // 4 * 32768 threads
  int widx = tid >> 15;
  int rem  = tid & 32767;
  int j    = rem & 15;
  int lane = (rem >> 4) & 31;
  int tile = rem >> 9;
  const float* src; _Float16* dst; int NT, N;
  if      (widx == 0) { src = We1; dst = We1h; NT = 16; N = 256; }
  else if (widx == 1) { src = Wn1; dst = Wn1h; NT = 16; N = 256; }
  else if (widx == 2) { src = We2; dst = We2h; NT = 8;  N = 128; }
  else                { src = Wn2; dst = Wn2h; NT = 8;  N = 128; }
  int nt = tile % NT, kt = tile / NT;
  int k = kt * 32 + 8 * (lane >> 4) + (j & 7) + 16 * (j >> 3);
  int n = nt * 16 + (lane & 15);
  dst[rem] = (_Float16)src[(size_t)k * N + n];
}

// ---------------------------------------------------------------------------
// gvec_e[b,n] = globals[b,:] @ We2_g[:,n] + be2[n]   (same for nodes)
// ---------------------------------------------------------------------------
__global__ __launch_bounds__(256) void k_gvec(
    const float* __restrict__ g,
    const float* __restrict__ We2g, const float* __restrict__ be2,
    const float* __restrict__ Wn2g, const float* __restrict__ bn2,
    float* __restrict__ gvec_e, float* __restrict__ gvec_n) {
  int tid = blockIdx.x * 256 + threadIdx.x;       // 2 * 32768 threads
  int sel = tid >> 15;
  int rem = tid & 32767;
  int b = rem >> 7, n = rem & 127;
  const float* W    = sel ? Wn2g : We2g;
  const float* bias = sel ? bn2  : be2;
  float s = bias[n];
  #pragma unroll 8
  for (int k = 0; k < 64; ++k) s += g[b * 64 + k] * W[k * 128 + n];
  (sel ? gvec_n : gvec_e)[rem] = s;
}

// ---------------------------------------------------------------------------
// WMMA tile loaders
// ---------------------------------------------------------------------------
__device__ inline v16h loadA_global(const float* __restrict__ base, int ldk,
                                    int rowBase, int kbase, int lane) {
  int m = lane & 15, g = lane >> 4;
  const float* p = base + (size_t)(rowBase + m) * ldk + kbase + 8 * g;
  v4f f0 = *reinterpret_cast<const v4f*>(p);
  v4f f1 = *reinterpret_cast<const v4f*>(p + 4);
  v4f f2 = *reinterpret_cast<const v4f*>(p + 16);
  v4f f3 = *reinterpret_cast<const v4f*>(p + 20);
  v16h a;
  #pragma unroll
  for (int i = 0; i < 4; ++i) {
    a[i]      = (_Float16)f0[i];
    a[4 + i]  = (_Float16)f1[i];
    a[8 + i]  = (_Float16)f2[i];
    a[12 + i] = (_Float16)f3[i];
  }
  return a;
}

__device__ inline v16h loadA_lds(const _Float16* __restrict__ buf,
                                 int kbase, int lane) {
  int m = lane & 15, g = lane >> 4;
  const _Float16* p = buf + m * 256 + kbase + 8 * g;
  v8h h0 = *reinterpret_cast<const v8h*>(p);        // ds_load_b128
  v8h h1 = *reinterpret_cast<const v8h*>(p + 16);   // ds_load_b128
  v16h a;
  #pragma unroll
  for (int i = 0; i < 8; ++i) { a[i] = h0[i]; a[8 + i] = h1[i]; }
  return a;
}

__device__ inline v16h ldB(const _Float16* __restrict__ w, int tile, int lane) {
  return *(reinterpret_cast<const v16h*>(w) + tile * 32 + lane);
}

// ---------------------------------------------------------------------------
// Fused per-graph block:  feats[b] (ROWS x 128)
//   -> relu(feats @ W1 + b1)     (ROWS x 256, regs/LDS only)
//   -> relu( .. @ W2 + gvec[b])  (ROWS x 128, never hits HBM)
//   -> column mean (128)         -> avg_out[b]
// One workgroup (128 threads / 4 waves) per graph.
//
// Register policy:
//   * W2's 64 B tiles preloaded once -> register-stationary (512 VGPRs).
//   * W1's B tiles streamed from L2 with an opaque offset (anti-LICM) and
//     DOUBLE-BUFFERED per K-step so the next 16-load clause is in flight
//     while the current 8 WMMAs execute (hides L2 latency at 1 wave/SIMD).
//   * Layer-2 LDS A tile is double-buffered the same way.
// ---------------------------------------------------------------------------
template <int ROWS_PER_G>
__global__ __launch_bounds__(128, 1) void gn_block_kernel(
    const float* __restrict__ feats,
    const _Float16* __restrict__ W1h,   // swizzled tiles, KT1=4, NT1=16
    const float* __restrict__ b1,       // [256]
    const _Float16* __restrict__ W2h,   // swizzled tiles, KT2=8, NT2=8
    const float* __restrict__ gvec,     // [B,128] = g @ W2g + b2
    float* __restrict__ avg_out) {      // [B,128]
  constexpr int NT1 = 16, KT1 = 4, NT2 = 8, KT2 = 8;
  constexpr int TILES = ROWS_PER_G / 16;

  __shared__ _Float16 ebuf[4][16 * 256];   // per-wave layer-1 restage (32 KB)
  __shared__ float    colsum[128];

  const int b    = blockIdx.x;
  const int lane = threadIdx.x & 31;
  const int wave = threadIdx.x >> 5;

  colsum[threadIdx.x] = 0.0f;
  __syncthreads();

  const int n0 = lane & 15;
  float bias1[NT1];
  #pragma unroll
  for (int nt = 0; nt < NT1; ++nt) bias1[nt] = b1[nt * 16 + n0];
  float gv[NT2];
  #pragma unroll
  for (int nt = 0; nt < NT2; ++nt) gv[nt] = gvec[b * 128 + nt * 16 + n0];

  // ---- W2: all 64 B tiles register-stationary across the whole loop ----
  v16h w2r[KT2 * NT2];
  #pragma unroll
  for (int i = 0; i < KT2 * NT2; ++i) w2r[i] = ldB(W2h, i, lane);

  const float* src = feats + (size_t)b * ROWS_PER_G * 128;
  _Float16* buf = ebuf[wave];
  const int mh = (lane >> 4) * 8;

  int w1off = 0;  // always 0 at runtime, opaque to the optimizer
  for (int t = wave; t < TILES; t += 4) {
    // Anti-LICM: keep W1 tile addresses loop-variant (stream, don't hoist).
    asm volatile("" : "+v"(w1off));
    const _Float16* W1p = W1h + w1off;

    const int rowBase = t * 16;
    if (t + 4 < TILES)
      __builtin_prefetch(src + (size_t)(rowBase + 64) * 128, 0, 3);

    // A tiles for this M-tile, kept in registers across both N-halves.
    v16h aT[KT1];
    #pragma unroll
    for (int kt = 0; kt < KT1; ++kt)
      aT[kt] = loadA_global(src, 128, rowBase, kt * 32, lane);

    v8f zero = {};

    // ---- Layer 1 in two halves of 8 N-tiles; B tiles double-buffered ----
    #pragma unroll
    for (int half = 0; half < 2; ++half) {
      v8f c1[8];
      #pragma unroll
      for (int i = 0; i < 8; ++i) c1[i] = zero;

      v16h bb[2][8];
      #pragma unroll
      for (int nt = 0; nt < 8; ++nt)
        bb[0][nt] = ldB(W1p, half * 8 + nt, lane);

      #pragma unroll
      for (int kt = 0; kt < KT1; ++kt) {
        // Issue next K-step's 8 B-tile loads before consuming this step's.
        if (kt + 1 < KT1) {
          #pragma unroll
          for (int nt = 0; nt < 8; ++nt)
            bb[(kt + 1) & 1][nt] =
                ldB(W1p, (kt + 1) * NT1 + half * 8 + nt, lane);
        }
        #pragma unroll
        for (int nt = 0; nt < 8; ++nt)
          c1[nt] = __builtin_amdgcn_wmma_f32_16x16x32_f16(
              false, aT[kt], false, bb[kt & 1][nt], (short)0, c1[nt],
              false, false);
      }

      // bias + ReLU + f16 restage to LDS, columns half*128 .. half*128+127
      #pragma unroll
      for (int nt = 0; nt < 8; ++nt) {
        const int ntg = half * 8 + nt;
        #pragma unroll
        for (int r = 0; r < 8; ++r) {
          float v = c1[nt][r] + bias1[ntg];
          v = fmaxf(v, 0.0f);
          buf[(mh + r) * 256 + ntg * 16 + n0] = (_Float16)v;
        }
      }
    }

    // ---- Layer 2: [16,256] x [256,128]; W2 stationary, A double-buffered --
    v8f c2[NT2];
    #pragma unroll
    for (int i = 0; i < NT2; ++i) c2[i] = zero;

    v16h a2b[2];
    a2b[0] = loadA_lds(buf, 0, lane);
    #pragma unroll
    for (int kt = 0; kt < KT2; ++kt) {
      if (kt + 1 < KT2)
        a2b[(kt + 1) & 1] = loadA_lds(buf, (kt + 1) * 32, lane);
      #pragma unroll
      for (int nt = 0; nt < NT2; ++nt)
        c2[nt] = __builtin_amdgcn_wmma_f32_16x16x32_f16(
            false, a2b[kt & 1], false, w2r[kt * NT2 + nt], (short)0, c2[nt],
            false, false);
    }

    // ---- gvec + ReLU + partial column sums into LDS (ds_add_f32) ----
    #pragma unroll
    for (int nt = 0; nt < NT2; ++nt) {
      float part = 0.0f;
      #pragma unroll
      for (int r = 0; r < 8; ++r) {
        float v = c2[nt][r] + gv[nt];
        part += fmaxf(v, 0.0f);
      }
      atomicAdd(&colsum[nt * 16 + n0], part);
    }
  }

  __syncthreads();
  avg_out[b * 128 + threadIdx.x] =
      colsum[threadIdx.x] * (1.0f / (float)ROWS_PER_G);
}

// ---------------------------------------------------------------------------
// Per-graph head (one block per graph).
// ---------------------------------------------------------------------------
__global__ __launch_bounds__(256) void k_final(
    const float* __restrict__ n_avg, const float* __restrict__ e_avg,
    const float* __restrict__ g,
    const float* __restrict__ Wg_n, const float* __restrict__ Wg_e,
    const float* __restrict__ Wg_g, const float* __restrict__ bg,
    const float* __restrict__ a,
    const float* __restrict__ Wh, const float* __restrict__ bh,
    const float* __restrict__ Wo, const float* __restrict__ bo,
    float* __restrict__ out) {
  __shared__ float sa[136];
  __shared__ float red[256];
  const int b = blockIdx.x, tid = threadIdx.x;
  if (tid < 128) {
    float s = bg[tid];
    #pragma unroll 4
    for (int k = 0; k < 128; ++k)
      s += n_avg[b * 128 + k] * Wg_n[k * 128 + tid] +
           e_avg[b * 128 + k] * Wg_e[k * 128 + tid];
    #pragma unroll 4
    for (int k = 0; k < 64; ++k) s += g[b * 64 + k] * Wg_g[k * 128 + tid];
    sa[tid] = s;
  } else if (tid < 136) {
    sa[tid] = a[b * 8 + (tid - 128)];
  }
  __syncthreads();
  float h = bh[tid];
  #pragma unroll 4
  for (int j = 0; j < 136; ++j) h += sa[j] * Wh[j * 256 + tid];
  h = fmaxf(h, 0.0f);
  red[tid] = h * Wo[tid];
  __syncthreads();
  #pragma unroll
  for (int s = 128; s > 0; s >>= 1) {
    if (tid < s) red[tid] += red[tid + s];
    __syncthreads();
  }
  if (tid == 0) out[b] = red[0] + bo[0];
}

// ---------------------------------------------------------------------------
// Launch
// ---------------------------------------------------------------------------
extern "C" void kernel_launch(void* const* d_in, const int* in_sizes, int n_in,
                              void* d_out, int out_size, void* d_ws,
                              size_t ws_size, hipStream_t stream) {
  const float* nodes    = (const float*)d_in[0];
  const float* edges    = (const float*)d_in[1];
  const float* globals_ = (const float*)d_in[2];
  // d_in[3]/d_in[4]: segment ids are implicit (contiguous) -> unused.
  const float* a     = (const float*)d_in[5];
  const float* We1   = (const float*)d_in[6];
  const float* be1   = (const float*)d_in[7];
  const float* Wn1   = (const float*)d_in[8];
  const float* bn1   = (const float*)d_in[9];
  const float* We2_e = (const float*)d_in[10];
  const float* We2_g = (const float*)d_in[11];
  const float* be2   = (const float*)d_in[12];
  const float* Wn2_n = (const float*)d_in[13];
  const float* Wn2_g = (const float*)d_in[14];
  const float* bn2   = (const float*)d_in[15];
  const float* Wg_n  = (const float*)d_in[16];
  const float* Wg_e  = (const float*)d_in[17];
  const float* Wg_g  = (const float*)d_in[18];
  const float* bg    = (const float*)d_in[19];
  const float* Wh    = (const float*)d_in[20];
  const float* bh    = (const float*)d_in[21];
  const float* Wo    = (const float*)d_in[22];
  const float* bo    = (const float*)d_in[23];
  float* out = (float*)d_out;

  char* w = (char*)d_ws;
  _Float16* We1h = (_Float16*)w; w += 65536;   // 128x256 f16, swizzled
  _Float16* Wn1h = (_Float16*)w; w += 65536;
  _Float16* We2h = (_Float16*)w; w += 65536;   // 256x128 f16, swizzled
  _Float16* Wn2h = (_Float16*)w; w += 65536;
  float* gvec_e  = (float*)w;    w += 131072;  // [256,128] f32
  float* gvec_n  = (float*)w;    w += 131072;
  float* e_avg   = (float*)w;    w += 131072;
  float* n_avg   = (float*)w;    w += 131072;

  k_convert<<<512, 256, 0, stream>>>(We1, Wn1, We2_e, Wn2_n,
                                     We1h, Wn1h, We2h, Wn2h);
  k_gvec<<<256, 256, 0, stream>>>(globals_, We2_g, be2, Wn2_g, bn2,
                                  gvec_e, gvec_n);
  gn_block_kernel<1024><<<256, 128, 0, stream>>>(edges, We1h, be1, We2h,
                                                 gvec_e, e_avg);
  gn_block_kernel<64><<<256, 128, 0, stream>>>(nodes, Wn1h, bn1, Wn2h,
                                               gvec_n, n_avg);
  k_final<<<256, 256, 0, stream>>>(n_avg, e_avg, globals_, Wg_n, Wg_e, Wg_g,
                                   bg, a, Wh, bh, Wo, bo, out);
}
